// OAVolumeRadiance_23888608100391
// MI455X (gfx1250) — matrix operand order repaired and checked by reading, not compiled, not run on hardware
//
#include <hip/hip_runtime.h>
#include <hip/hip_bf16.h>
#include <math.h>

// CDNA5 (gfx1250) fully-fused tcnn-style radiance MLP:
//   [feat16 | SH16(refl dirs) | normals3] (pad 64) -> 64 relu -> 64 relu -> 3 sigmoid
// One wave = 32 samples/iter (two 16-row WMMA tiles). Weights register-resident
// as v_wmma_f32_16x16x32_f16 B-fragments; activations bounce through per-wave LDS.

typedef __attribute__((ext_vector_type(16))) _Float16 v16h;
typedef __attribute__((ext_vector_type(8)))  _Float16 v8h;
typedef __attribute__((ext_vector_type(8)))  float    v8f;
typedef __attribute__((ext_vector_type(4)))  float    v4f;

#define WAVES_PER_BLOCK 4
#define BLOCK_THREADS   128

__device__ __forceinline__ void lds_fence() {
  // Same-wave DS ops are in-order on CDNA5; this is a compiler fence plus a
  // hard wait so nothing is scheduled across the produce->consume boundary.
  asm volatile("s_wait_dscnt 0" ::: "memory");
}

__device__ __forceinline__ float sigmoidf(float x) {
  return 1.0f / (1.0f + __expf(-x));
}

// ReLU in the f16 domain: v_cvt_f16_f32 output is canonical, so maxnum needs
// no input-canonicalize -> single v_max_num_f16 (pairable into v_pk_max_num_f16).
__device__ __forceinline__ _Float16 relu_cvt(float x) {
  return __builtin_fmaxf16((_Float16)x, (_Float16)0.0f);
}

__global__ void __launch_bounds__(BLOCK_THREADS)
radiance_mlp_fused(const float* __restrict__ feat,
                   const float* __restrict__ dirs,
                   const float* __restrict__ norms,
                   const float* __restrict__ W1,   // [35][64] row-major
                   const float* __restrict__ W2,   // [64][64]
                   const float* __restrict__ W3,   // [64][3]
                   float* __restrict__ out,        // [N][3]
                   int nSamples)
{
  __shared__ __align__(16) _Float16 sX[WAVES_PER_BLOCK][32 * 64];
  __shared__ __align__(16) _Float16 sH[WAVES_PER_BLOCK][32 * 64];
  __shared__ __align__(16) float    sO[WAVES_PER_BLOCK][96];

  const int lane = threadIdx.x & 31;
  const int wib  = threadIdx.x >> 5;
  const int half = lane >> 4;   // 0: lanes 0-15, 1: lanes 16-31
  const int lrow = lane & 15;

  _Float16* Xw = sX[wib];
  _Float16* Hw = sH[wib];
  float*    Ow = sO[wib];

  // ---------------- Weight B-fragments (gathered once per wave) ----------------
  // v_wmma_f32_16x16x32_f16 B (32x16, KxN) layout: lane holds column n=lane%16;
  // this code puts K=16*half+j in element j (contiguous k order, not AMD's table order).
  v16h w1f[2][4], w2f[2][4], w3f[2];
#pragma unroll
  for (int ks = 0; ks < 2; ++ks) {
    const int kb = ks * 32 + half * 16;
#pragma unroll
    for (int nt = 0; nt < 4; ++nt) {
      const int n = nt * 16 + lrow;
      v16h f1, f2;
#pragma unroll
      for (int j = 0; j < 16; ++j) {
        const int k = kb + j;
        f1[j] = (_Float16)((k < 35) ? W1[k * 64 + n] : 0.0f);  // zero-pad K 35..63
        f2[j] = (_Float16)(W2[k * 64 + n]);
      }
      w1f[ks][nt] = f1;
      w2f[ks][nt] = f2;
    }
    {
      const int n = lrow;
      v16h f3;
#pragma unroll
      for (int j = 0; j < 16; ++j) {
        const int k = kb + j;
        f3[j] = (_Float16)((n < 3) ? W3[k * 3 + n] : 0.0f);    // zero-pad N 3..15
      }
      w3f[ks] = f3;
    }
  }

  // A-fragment loader (16x32 f16 A layout): lane = row lrow; element j from
  // K = ks*32 + 16*half + j (matches B fragments).
  auto load_a = [&](const _Float16* tileBase, int ks) -> v16h {
    const _Float16* rp = tileBase + lrow * 64 + ks * 32 + half * 16;
    v8h lo = *(const v8h*)(rp);
    v8h hi = *(const v8h*)(rp + 8);
    v16h a;
#pragma unroll
    for (int i = 0; i < 8; ++i) { a[i] = lo[i]; a[i + 8] = hi[i]; }
    return a;
  };

  const int nChunks  = (nSamples + 31) >> 5;
  const int totWaves = (gridDim.x * blockDim.x) >> 5;
  const int gw       = (blockIdx.x * blockDim.x + threadIdx.x) >> 5;

  for (int c = gw; c < nChunks; c += totWaves) {
    const int base = c << 5;
    const int s    = base + lane;

    // -------- per-lane input assembly: features | SH(deg4, reflected) | normals --------
    v8h r[8] = {};
    if (s < nSamples) {
      const v4f* f4 = (const v4f*)(feat + (size_t)s * 16);
      v4f a0 = __builtin_nontemporal_load(f4 + 0);
      v4f a1 = __builtin_nontemporal_load(f4 + 1);
      v4f a2 = __builtin_nontemporal_load(f4 + 2);
      v4f a3 = __builtin_nontemporal_load(f4 + 3);
      float fv[16] = {a0[0], a0[1], a0[2], a0[3], a1[0], a1[1], a1[2], a1[3],
                      a2[0], a2[1], a2[2], a2[3], a3[0], a3[1], a3[2], a3[3]};

      const float* dp = dirs  + 3 * (size_t)s;
      const float* np = norms + 3 * (size_t)s;
      const float dx = __builtin_nontemporal_load(dp + 0);
      const float dy = __builtin_nontemporal_load(dp + 1);
      const float dz = __builtin_nontemporal_load(dp + 2);
      const float nx = __builtin_nontemporal_load(np + 0);
      const float ny = __builtin_nontemporal_load(np + 1);
      const float nz = __builtin_nontemporal_load(np + 2);

      const float dot = dx * nx + dy * ny + dz * nz;
      const float rx = dx - 2.0f * dot * nx;
      const float ry = dy - 2.0f * dot * ny;
      const float rz = dz - 2.0f * dot * nz;

      const float xy = rx * ry, xz = rx * rz, yz = ry * rz;
      const float x2 = rx * rx, y2 = ry * ry, z2 = rz * rz;
      float sh[16];
      sh[0]  = 0.28209479177387814f;
      sh[1]  = -0.48860251190291987f * ry;
      sh[2]  =  0.48860251190291987f * rz;
      sh[3]  = -0.48860251190291987f * rx;
      sh[4]  =  1.0925484305920792f * xy;
      sh[5]  = -1.0925484305920792f * yz;
      sh[6]  =  0.94617469575755997f * z2 - 0.31539156525251999f;
      sh[7]  = -1.0925484305920792f * xz;
      sh[8]  =  0.54627421529603959f * (x2 - y2);
      sh[9]  =  0.59004358992664352f * ry * (-3.0f * x2 + y2);
      sh[10] =  2.8906114426405538f * xy * rz;
      sh[11] =  0.45704579946446572f * ry * (1.0f - 5.0f * z2);
      sh[12] =  0.3731763325901154f * rz * (5.0f * z2 - 3.0f);
      sh[13] =  0.45704579946446572f * rx * (1.0f - 5.0f * z2);
      sh[14] =  1.4453057213202769f * rz * (x2 - y2);
      sh[15] =  0.59004358992664352f * rx * (-x2 + 3.0f * y2);

#pragma unroll
      for (int i = 0; i < 16; ++i) r[i >> 3][i & 7]       = (_Float16)fv[i];
#pragma unroll
      for (int i = 0; i < 16; ++i) r[2 + (i >> 3)][i & 7] = (_Float16)sh[i];
      r[4][0] = (_Float16)nx; r[4][1] = (_Float16)ny; r[4][2] = (_Float16)nz;
    }
#pragma unroll
    for (int j = 0; j < 8; ++j)
      *(v8h*)(Xw + lane * 64 + j * 8) = r[j];
    lds_fence();

    // -------- two 16-row tiles through the 3 layers --------
#pragma unroll
    for (int t = 0; t < 2; ++t) {
      const _Float16* Xt = Xw + t * 16 * 64;
      _Float16*       Ht = Hw + t * 16 * 64;

      // Layer 1: [16x64] x [64(pad)x64]
      v8f acc1[4] = {};
#pragma unroll
      for (int ks = 0; ks < 2; ++ks) {
        v16h A = load_a(Xt, ks);
#pragma unroll
        for (int nt = 0; nt < 4; ++nt)
          acc1[nt] = __builtin_amdgcn_wmma_f32_16x16x32_f16(
              false, A, false, w1f[ks][nt], (short)0, acc1[nt], false, false);
      }
      // cvt -> f16 ReLU -> LDS (D layout: VGPR v holds row v+8*half, col lrow)
#pragma unroll
      for (int nt = 0; nt < 4; ++nt)
#pragma unroll
        for (int v = 0; v < 8; ++v)
          Ht[(v + half * 8) * 64 + nt * 16 + lrow] = relu_cvt(acc1[nt][v]);
      lds_fence();

      // Layer 2: [16x64] x [64x64]
      v8f acc2[4] = {};
#pragma unroll
      for (int ks = 0; ks < 2; ++ks) {
        v16h A = load_a(Ht, ks);
#pragma unroll
        for (int nt = 0; nt < 4; ++nt)
          acc2[nt] = __builtin_amdgcn_wmma_f32_16x16x32_f16(
              false, A, false, w2f[ks][nt], (short)0, acc2[nt], false, false);
      }
      // cvt -> f16 ReLU -> reuse X buffer (this tile's X rows are consumed)
      _Float16* H2t = Xw + t * 16 * 64;
#pragma unroll
      for (int nt = 0; nt < 4; ++nt)
#pragma unroll
        for (int v = 0; v < 8; ++v)
          H2t[(v + half * 8) * 64 + nt * 16 + lrow] = relu_cvt(acc2[nt][v]);
      lds_fence();

      // Layer 3: [16x64] x [64x16(pad)] -> sigmoid on cols 0..2
      v8f acc3 = {};
#pragma unroll
      for (int ks = 0; ks < 2; ++ks) {
        v16h A = load_a(H2t, ks);
        acc3 = __builtin_amdgcn_wmma_f32_16x16x32_f16(
            false, A, false, w3f[ks], (short)0, acc3, false, false);
      }
      if (lrow < 3) {
#pragma unroll
        for (int v = 0; v < 8; ++v)
          Ow[t * 48 + (v + half * 8) * 3 + lrow] = sigmoidf(acc3[v]);
      }
    }
    lds_fence();

    // -------- coalesced output: 96 floats (32 samples x RGB), streaming store --------
    if (s < nSamples) {
      float* op = out + (size_t)base * 3 + 3 * lane;
      __builtin_nontemporal_store(Ow[3 * lane + 0], op + 0);
      __builtin_nontemporal_store(Ow[3 * lane + 1], op + 1);
      __builtin_nontemporal_store(Ow[3 * lane + 2], op + 2);
    }
    // No trailing fence needed: next write to Ow/Xw is separated by three
    // fences in the following iteration, and same-wave DS ops are in-order.
  }
}

extern "C" void kernel_launch(void* const* d_in, const int* in_sizes, int n_in,
                              void* d_out, int out_size, void* d_ws, size_t ws_size,
                              hipStream_t stream) {
  const float* feat  = (const float*)d_in[0];  // [N,16]
  const float* dirs  = (const float*)d_in[1];  // [N,3]
  const float* norms = (const float*)d_in[2];  // [N,3]
  const float* W1    = (const float*)d_in[3];  // [35,64]
  const float* W2    = (const float*)d_in[4];  // [64,64]
  const float* W3    = (const float*)d_in[5];  // [64,3]
  float* out = (float*)d_out;

  const int nSamples = in_sizes[0] / 16;
  const int blocks = 2048;  // 8192 waves; 2M samples -> 8 chunks of 32 per wave
  radiance_mlp_fused<<<blocks, BLOCK_THREADS, 0, stream>>>(
      feat, dirs, norms, W1, W2, W3, out, nSamples);
}
